// BitShiftMamba_27599459844412
// MI455X (gfx1250) — matrix operand order repaired and checked by reading, not compiled, not run on hardware
//
#include <hip/hip_runtime.h>
#include <hip/hip_bf16.h>
#include <stdint.h>

// Model dims (compile-time constants from the reference)
#define B_   8
#define L_   1024
#define DM_  512
#define DS_  16
#define DC_  4
#define NL_  2
#define DI_  1024            // EXP*DM
#define DR_  32              // DM/16
#define DXB_ (DR_ + 2*DS_)   // 64 cols of xdbl

typedef __attribute__((ext_vector_type(16))) __bf16 v16bf;
typedef __attribute__((ext_vector_type(8)))  __bf16 v8bf;
typedef __attribute__((ext_vector_type(8)))  float  v8f;
typedef __attribute__((ext_vector_type(4)))  uint32_t v4u;
typedef __attribute__((ext_vector_type(4)))  int      v4i;
typedef __attribute__((ext_vector_type(8)))  int      v8i;

#if defined(__has_builtin)
#if __has_builtin(__builtin_amdgcn_tensor_load_to_lds)
#define HAVE_TDM 1
#endif
#endif

__device__ __forceinline__ __bf16 f2bf(float x) {
  union { float f; uint32_t u; } v; v.f = x;
  uint32_t r = (v.u + 0x7FFFu + ((v.u >> 16) & 1u)) >> 16;   // round-to-nearest-even
  union { uint16_t s; __bf16 b; } o; o.s = (uint16_t)r;
  return o.b;
}
__device__ __forceinline__ float bf2f(__bf16 b) {
  union { uint16_t s; __bf16 b; } i; i.b = b;
  union { uint32_t u; float f; } o; o.u = ((uint32_t)i.s) << 16;
  return o.f;
}

// ---------------------------------------------------------------------------
// f32 -> bf16 bulk convert (weights, once per launch)
// ---------------------------------------------------------------------------
__global__ void cvt_bf16_kernel(const float* __restrict__ in, __bf16* __restrict__ out, int n) {
  int i = blockIdx.x * blockDim.x + threadIdx.x;
  if (i < n) out[i] = f2bf(in[i]);
}

// ---------------------------------------------------------------------------
// Embedding gather -> bf16 activations: h[b,l,:] = emb[x[b,l],:]
// ---------------------------------------------------------------------------
__global__ void embed_kernel(const int* __restrict__ x, const float* __restrict__ emb,
                             __bf16* __restrict__ hbf, int n) {
  int i = blockIdx.x * blockDim.x + threadIdx.x;
  if (i >= n) return;
  int row = i / DM_;
  int c   = i - row * DM_;
  hbf[i] = f2bf(emb[x[row] * DM_ + c]);
}

// ---------------------------------------------------------------------------
// bf16 WMMA GEMM: C[M,N] = A[M,K] * W[N,K]^T   (bf16 in, fp32 acc)
// Block tile 64x64x32, 128 threads = 4 waves (2x2), each wave 32x32 = 4 WMMAs.
// A tile staged via global_load_async_to_lds_b128 (ASYNCcnt, per-lane chunks).
// W tile staged via TENSOR_LOAD_TO_LDS (TDM, TENSORcnt): one D# per tile, HW
// padding reproduces the 80B LDS row stride (16 DW rows + 4 DW pad).
// Double-buffered software pipeline.
// ---------------------------------------------------------------------------
#define BM 64
#define BN 64
#define BK 32
#define GEMM_THREADS 128
#define LDT (BK + 8)   // padded bf16 row stride (80B: keeps 16B-aligned subrows)

__device__ __forceinline__ void tdm_load_tile_64x32_bf16(uint32_t lds_addr,
                                                         const void* gaddr,
                                                         int tensor_d0,
                                                         int d0_stride)
{
#ifdef HAVE_TDM
  uint64_t ga = (uint64_t)(uintptr_t)gaddr;
  v4u g0;
  g0[0] = 1u;                                   // count=1, user load descriptor
  g0[1] = lds_addr;                             // LDS byte address of tile
  g0[2] = (uint32_t)ga;                         // global_addr[31:0]
  g0[3] = (uint32_t)(ga >> 32) | (2u << 30);    // global_addr[56:32] | type=2

  v8i g1;
  // pad_amount=3 (4 DW = 16B) [31:25] | pad_interval=3 (16 DW = 64B) [24:22]
  // | pad_enable [20] | data_size=1 (2B) [17:16] ; workgroup_mask=0
  g1[0] = (int)((3u << 25) | (3u << 22) | (1u << 20) | (1u << 16));
  g1[1] = (int)(((uint32_t)tensor_d0 & 0xFFFFu) << 16);   // tensor_dim0 lo16
  g1[2] = (int)(((uint32_t)tensor_d0 >> 16) | (0xFFFFu << 16)); // dim0 hi16 | dim1 lo16
  g1[3] = (int)(0u | (32u << 16));              // dim1 hi16=0 ; tile_dim0=32
  g1[4] = (int)(64u);                           // tile_dim1=64 ; tile_dim2=0
  g1[5] = (int)d0_stride;                       // tensor_dim0_stride lo32 (elems)
  g1[6] = 0;                                    // stride hi16 | dim1_stride lo16
  g1[7] = 0;                                    // dim1_stride hi32
  v4i z4 = {0, 0, 0, 0};
#if __clang_major__ >= 23
  v8i z8 = {0, 0, 0, 0, 0, 0, 0, 0};
  __builtin_amdgcn_tensor_load_to_lds(g0, g1, z4, z4, z8, 0);
#else
  __builtin_amdgcn_tensor_load_to_lds(g0, g1, z4, z4, 0);
#endif
#else
  (void)lds_addr; (void)gaddr; (void)tensor_d0; (void)d0_stride;
#endif
}

__global__ __launch_bounds__(GEMM_THREADS)
void gemm_bf16_wmma(const __bf16* __restrict__ A, int lda,
                    const __bf16* __restrict__ W, int ldw,
                    float* __restrict__ C, int ldc,
                    __bf16* __restrict__ Cbf, int ldcb,
                    int K)
{
  __shared__ __bf16 As[2][BM][LDT];
  __shared__ __bf16 Ws[2][BN][LDT];

  const int tid  = threadIdx.x;
  const int lane = tid & 31;
  const int wave = tid >> 5;
  const int wm   = (wave >> 1) * 32;
  const int wn   = (wave & 1)  * 32;
  const long bm  = (long)blockIdx.y * BM;
  const long bn  = (long)blockIdx.x * BN;

  const int row  = lane & 15;   // 0..15
  const int half = lane >> 4;   // 0..1

  v8f acc[2][2] = {};

  // A tile: 64 rows x 4 chunks(16B) = 256 chunks; 2 async copies per thread.
  auto issue_A = [&](int kt, int buf) {
    const int k0 = kt * BK;
    #pragma unroll
    for (int it = 0; it < 2; ++it) {
      int idx = tid + it * GEMM_THREADS;       // 0..255
      int r   = idx >> 2;
      int c8  = (idx & 3) << 3;                // bf16 elem offset of 16B chunk
      const __bf16* ga = A + (bm + r) * (long)lda + k0 + c8;
      uint32_t la = (uint32_t)(uintptr_t)(void*)&As[buf][r][c8];
      asm volatile("global_load_async_to_lds_b128 %0, %1, off"
                   :: "v"(la), "v"(ga) : "memory");
    }
  };
  // W tile: one TDM descriptor (wave 0 only); HW pads rows to the LDS stride.
  auto issue_W = [&](int kt, int buf) {
#ifdef HAVE_TDM
    if (wave == 0) {
      const __bf16* gw = W + bn * (long)ldw + kt * BK;
      tdm_load_tile_64x32_bf16((uint32_t)(uintptr_t)(void*)&Ws[buf][0][0],
                               (const void*)gw, K, ldw);
    }
#else
    const int k0 = kt * BK;
    #pragma unroll
    for (int it = 0; it < 2; ++it) {
      int idx = tid + it * GEMM_THREADS;
      int r   = idx >> 2;
      int c8  = (idx & 3) << 3;
      const __bf16* gw = W + (bn + r) * (long)ldw + k0 + c8;
      uint32_t lw = (uint32_t)(uintptr_t)(void*)&Ws[buf][r][c8];
      asm volatile("global_load_async_to_lds_b128 %0, %1, off"
                   :: "v"(lw), "v"(gw) : "memory");
    }
#endif
  };

  const int T = K / BK;
  issue_A(0, 0); issue_W(0, 0);
  if (T > 1) { issue_A(1, 1); issue_W(1, 1); }

  for (int kt = 0; kt < T; ++kt) {
    const int buf = kt & 1;
    // Async/tensor ops complete in order per wave: waiting down to one tile's
    // worth of outstanding ops means tile kt is resident in LDS.
    if (kt + 1 < T) {
#ifdef HAVE_TDM
      asm volatile("s_wait_asynccnt 0x2" ::: "memory");
      __builtin_amdgcn_s_wait_tensorcnt(1);
#else
      asm volatile("s_wait_asynccnt 0x4" ::: "memory");
#endif
    } else {
      asm volatile("s_wait_asynccnt 0x0" ::: "memory");
#ifdef HAVE_TDM
      __builtin_amdgcn_s_wait_tensorcnt(0);
#endif
    }
    __syncthreads();

    // A fragment (16-bit A 16x32): lanes 0-15 hold K{0..7,16..23},
    // lanes 16-31 hold K{8..15,24..31} of row (lane&15).
    v16bf afrag[2], bfrag[2];
    #pragma unroll
    for (int mi = 0; mi < 2; ++mi) {
      const __bf16* ap = &As[buf][wm + mi * 16 + row][half * 8];
      v8bf lo = *(const v8bf*)ap;
      v8bf hi = *(const v8bf*)(ap + 16);
      #pragma unroll
      for (int j = 0; j < 8; ++j) { afrag[mi][j] = lo[j]; afrag[mi][8 + j] = hi[j]; }
    }
    // B fragment (16-bit B 32x16): lanes 0-15 hold K{0..7,16..23}, lanes 16-31
    // K{8..15,24..31} of column (lane&15), matching the A fragment.
    #pragma unroll
    for (int ni = 0; ni < 2; ++ni) {
      const __bf16* bp = &Ws[buf][wn + ni * 16 + row][half * 8];
      v8bf lo = *(const v8bf*)bp;
      v8bf hi = *(const v8bf*)(bp + 16);
      #pragma unroll
      for (int j = 0; j < 8; ++j) { bfrag[ni][j] = lo[j]; bfrag[ni][8 + j] = hi[j]; }
    }

    #pragma unroll
    for (int mi = 0; mi < 2; ++mi)
      #pragma unroll
      for (int ni = 0; ni < 2; ++ni)
        acc[mi][ni] = __builtin_amdgcn_wmma_f32_16x16x32_bf16(
            false, afrag[mi], false, bfrag[ni], (short)0, acc[mi][ni], false, false);

    __syncthreads();   // everyone done reading buf before it is refilled
    if (kt + 2 < T) { issue_A(kt + 2, buf); issue_W(kt + 2, buf); }
  }

  // D layout: VGPR r -> M = r + 8*(lane>>4), N = lane&15
  #pragma unroll
  for (int mi = 0; mi < 2; ++mi)
    #pragma unroll
    for (int ni = 0; ni < 2; ++ni) {
      long gm = bm + wm + mi * 16 + half * 8;
      long gn = bn + wn + ni * 16 + row;
      #pragma unroll
      for (int r = 0; r < 8; ++r)
        C[(gm + r) * (long)ldc + gn] = acc[mi][ni][r];
      if (Cbf != nullptr) {
        #pragma unroll
        for (int r = 0; r < 8; ++r)
          Cbf[(gm + r) * (long)ldcb + gn] = f2bf(acc[mi][ni][r]);
      }
    }
}

// ---------------------------------------------------------------------------
// Causal depthwise conv (DC=4) + bias + SiLU -> bf16 u
// ---------------------------------------------------------------------------
__global__ void conv_silu_kernel(const float* __restrict__ xz,
                                 const float* __restrict__ cw,
                                 const float* __restrict__ cb,
                                 __bf16* __restrict__ u, int n)
{
  int i = blockIdx.x * blockDim.x + threadIdx.x;
  if (i >= n) return;
  int d  = i % DI_;
  int bt = i / DI_;            // b*L + t
  int t  = bt % L_;
  int b0 = bt - t;             // b*L
  float acc = cb[d];
  #pragma unroll
  for (int k = 0; k < DC_; ++k) {
    int ts = t + k - (DC_ - 1);
    if (ts >= 0)
      acc += xz[(long)(b0 + ts) * (2 * DI_) + d] * cw[d * DC_ + k];
  }
  u[i] = f2bf(acc / (1.f + __expf(-acc)));   // silu
}

// ---------------------------------------------------------------------------
// Selective scan: one block per batch, one thread per channel d (DI=1024),
// 16 states in registers; B/C vectors chunk-staged through LDS (broadcast).
// Fuses softplus(delta), D skip, silu(z) gating. Writes yg (B*L, DI) bf16.
// ---------------------------------------------------------------------------
#define SCAN_CH 64
__global__ __launch_bounds__(1024)
void scan_kernel(const float* __restrict__ xdbl,    // (B*L, 64): dt|B|C
                 const float* __restrict__ dtproj,  // (B*L, DI)
                 const __bf16* __restrict__ ubuf,   // (B*L, DI)
                 const float* __restrict__ xz,      // (B*L, 2*DI): z upper half
                 const float* __restrict__ A_log,   // (DI, DS) layer slice
                 const float* __restrict__ bdt,     // (DI)
                 const float* __restrict__ Dp,      // (DI)
                 __bf16* __restrict__ yg)           // (B*L, DI)
{
  __shared__ float bc[SCAN_CH][2 * DS_];
  const int b = blockIdx.x;
  const int d = threadIdx.x;

  float a[DS_], st[DS_];
  #pragma unroll
  for (int s = 0; s < DS_; ++s) { a[s] = -__expf(A_log[d * DS_ + s]); st[s] = 0.f; }
  const float bd = bdt[d], dp = Dp[d];

  for (int c = 0; c < L_ / SCAN_CH; ++c) {
    for (int i = d; i < SCAN_CH * 2 * DS_; i += 1024) {
      int tt = i >> 5, j = i & 31;
      bc[tt][j] = xdbl[(long)(b * L_ + c * SCAN_CH + tt) * DXB_ + DR_ + j];
    }
    __syncthreads();
    for (int tt = 0; tt < SCAN_CH; ++tt) {
      long r = (long)(b * L_ + c * SCAN_CH + tt);
      float xv = dtproj[r * DI_ + d] + bd;
      float delta = (xv > 20.f) ? xv : log1pf(__expf(xv));
      float ut = bf2f(ubuf[r * DI_ + d]);
      float du = delta * ut;
      float y = 0.f;
      #pragma unroll
      for (int s = 0; s < DS_; ++s) {
        st[s] = st[s] * __expf(delta * a[s]) + du * bc[tt][s];
        y += st[s] * bc[tt][DS_ + s];
      }
      y += ut * dp;
      float z = xz[r * (2 * DI_) + DI_ + d];
      yg[r * DI_ + d] = f2bf(y * (z / (1.f + __expf(-z))));
    }
    __syncthreads();
  }
}

// ---------------------------------------------------------------------------
// LayerNorm + head: one wave32 per row of DM=512, shuffle reductions.
// ---------------------------------------------------------------------------
__global__ __launch_bounds__(256)
void ln_head_kernel(const float* __restrict__ h,
                    const float* __restrict__ ln_g, const float* __restrict__ ln_b,
                    const float* __restrict__ head_w, const float* __restrict__ head_b,
                    float* __restrict__ out)
{
  const int lane = threadIdx.x & 31;
  const int wave = threadIdx.x >> 5;
  const long row = (long)blockIdx.x * 8 + wave;
  const float* hr = h + row * DM_;

  float s = 0.f, s2 = 0.f;
  for (int c = lane; c < DM_; c += 32) { float v = hr[c]; s += v; s2 += v * v; }
  #pragma unroll
  for (int m = 16; m; m >>= 1) { s += __shfl_xor(s, m, 32); s2 += __shfl_xor(s2, m, 32); }
  float mu = s * (1.f / DM_);
  float rstd = rsqrtf(s2 * (1.f / DM_) - mu * mu + 1e-5f);

  float p = 0.f;
  for (int c = lane; c < DM_; c += 32) {
    float hn = (hr[c] - mu) * rstd * ln_g[c] + ln_b[c];
    p += hn * head_w[c];
  }
  #pragma unroll
  for (int m = 16; m; m >>= 1) p += __shfl_xor(p, m, 32);
  if (lane == 0) out[row] = p + head_b[0];
}

// ---------------------------------------------------------------------------
extern "C" void kernel_launch(void* const* d_in, const int* in_sizes, int n_in,
                              void* d_out, int out_size, void* d_ws, size_t ws_size,
                              hipStream_t stream)
{
  (void)in_sizes; (void)n_in; (void)out_size; (void)ws_size;
  const int*   x     = (const int*)  d_in[0];
  const float* emb   = (const float*)d_in[1];
  const float* Win   = (const float*)d_in[2];
  const float* convw = (const float*)d_in[3];
  const float* convb = (const float*)d_in[4];
  const float* Wx    = (const float*)d_in[5];
  const float* Wdt   = (const float*)d_in[6];
  const float* bdt   = (const float*)d_in[7];
  const float* A_log = (const float*)d_in[8];
  const float* Dp    = (const float*)d_in[9];
  const float* Wout  = (const float*)d_in[10];
  const float* ln_g  = (const float*)d_in[11];
  const float* ln_b  = (const float*)d_in[12];
  const float* hw    = (const float*)d_in[13];
  const float* hb    = (const float*)d_in[14];
  float* out = (float*)d_out;

  const size_t nBL = (size_t)B_ * L_;          // 8192 rows
  char* p = (char*)d_ws;
  auto carve = [&](size_t bytes) -> void* {
    void* r = (void*)p; p += (bytes + 255) & ~(size_t)255; return r;
  };
  float*  hF      = (float*) carve(nBL * DM_ * 4);          // fp32 h (LN input)
  __bf16* hbf     = (__bf16*)carve(nBL * DM_ * 2);          // bf16 h (GEMM1 input)
  float*  xz      = (float*) carve(nBL * 2 * DI_ * 4);
  __bf16* ub      = (__bf16*)carve(nBL * DI_ * 2);
  float*  xdbl    = (float*) carve(nBL * DXB_ * 4);
  __bf16* xdbl_bf = (__bf16*)carve(nBL * DXB_ * 2);
  float*  dtp     = (float*) carve(nBL * DI_ * 4);
  __bf16* yg      = (__bf16*)carve(nBL * DI_ * 2);
  __bf16* Winb    = (__bf16*)carve((size_t)NL_ * 2 * DI_ * DM_ * 2);
  __bf16* Wxb     = (__bf16*)carve((size_t)NL_ * DXB_ * DI_ * 2);
  __bf16* Wdtb    = (__bf16*)carve((size_t)NL_ * DI_ * DR_ * 2);
  __bf16* Woutb   = (__bf16*)carve((size_t)NL_ * DM_ * DI_ * 2);

  // Weight conversion (f32 -> bf16), once per launch
  {
    int n1 = NL_ * 2 * DI_ * DM_;
    cvt_bf16_kernel<<<(n1 + 255) / 256, 256, 0, stream>>>(Win, Winb, n1);
    int n2 = NL_ * DXB_ * DI_;
    cvt_bf16_kernel<<<(n2 + 255) / 256, 256, 0, stream>>>(Wx, Wxb, n2);
    int n3 = NL_ * DI_ * DR_;
    cvt_bf16_kernel<<<(n3 + 255) / 256, 256, 0, stream>>>(Wdt, Wdtb, n3);
    int n4 = NL_ * DM_ * DI_;
    cvt_bf16_kernel<<<(n4 + 255) / 256, 256, 0, stream>>>(Wout, Woutb, n4);
  }

  {
    int n = (int)(nBL * DM_);
    embed_kernel<<<(n + 255) / 256, 256, 0, stream>>>(x, emb, hbf, n);
  }

  for (int i = 0; i < NL_; ++i) {
    const __bf16* Win_i  = Winb  + (size_t)i * 2 * DI_ * DM_;
    const float*  cw_i   = convw + (size_t)i * DI_ * DC_;
    const float*  cb_i   = convb + (size_t)i * DI_;
    const __bf16* Wx_i   = Wxb   + (size_t)i * DXB_ * DI_;
    const __bf16* Wdt_i  = Wdtb  + (size_t)i * DI_ * DR_;
    const float*  bdt_i  = bdt   + (size_t)i * DI_;
    const float*  Al_i   = A_log + (size_t)i * DI_ * DS_;
    const float*  Dp_i   = Dp    + (size_t)i * DI_;
    const __bf16* Wout_i = Woutb + (size_t)i * DM_ * DI_;

    // xz = h @ Win^T   (M=8192, N=2048, K=512)
    gemm_bf16_wmma<<<dim3((2 * DI_) / BN, (B_ * L_) / BM), GEMM_THREADS, 0, stream>>>(
        hbf, DM_, Win_i, DM_, xz, 2 * DI_, nullptr, 0, DM_);

    { int n = (int)(nBL * DI_);
      conv_silu_kernel<<<(n + 255) / 256, 256, 0, stream>>>(xz, cw_i, cb_i, ub, n); }

    // xdbl = u @ Wx^T  (N=64, K=1024); also shadow-store bf16 for the delta GEMM
    gemm_bf16_wmma<<<dim3(DXB_ / BN, (B_ * L_) / BM), GEMM_THREADS, 0, stream>>>(
        ub, DI_, Wx_i, DI_, xdbl, DXB_, xdbl_bf, DXB_, DI_);

    // dtproj = dt @ Wdt^T  (N=1024, K=32; A = first 32 cols of xdbl_bf, lda=64)
    gemm_bf16_wmma<<<dim3(DI_ / BN, (B_ * L_) / BM), GEMM_THREADS, 0, stream>>>(
        xdbl_bf, DXB_, Wdt_i, DR_, dtp, DI_, nullptr, 0, DR_);

    scan_kernel<<<B_, DI_, 0, stream>>>(xdbl, dtp, ub, xz, Al_i, bdt_i, Dp_i, yg);

    // h' = yg @ Wout^T  (N=512, K=1024); fp32 for LN + bf16 for next layer GEMM1
    gemm_bf16_wmma<<<dim3(DM_ / BN, (B_ * L_) / BM), GEMM_THREADS, 0, stream>>>(
        yg, DI_, Wout_i, DI_, hF, DM_, hbf, DM_, DI_);
  }

  ln_head_kernel<<<(B_ * L_) / 8, 256, 0, stream>>>(hF, ln_g, ln_b, hw, hb, out);
}
